// LSTMEncoderDecoder_50697793962489
// MI455X (gfx1250) — compile-verified, matrix operand order checked by reading, not run on hardware
//
#include <hip/hip_runtime.h>
#include <hip/hip_bf16.h>
#include <stdint.h>
#include <stddef.h>

// ---------------------------------------------------------------------------
// LSTM encoder-decoder for MI455X (gfx1250), bf16 WMMA path.
//   B=512, T=82, D=129, H=1024, L=2, OUT=128, decoder steps = 100
// Gate GEMMs: each wave computes a 32x32 f32 tile via 4 accumulators of
// v_wmma_f32_16x16x32_bf16, with A/B operand reuse-hint chaining.
// ---------------------------------------------------------------------------

typedef __bf16 bf16_t;
typedef bf16_t v16bf __attribute__((ext_vector_type(16)));
typedef float  v8f   __attribute__((ext_vector_type(8)));

#define BATCH   512
#define TSTEPS  82
#define DSTEPS  100
#define DIN     129
#define KP0     160      // DIN padded to multiple of 32
#define HB      1024
#define NG      4096     // 4*H gate width
#define OUTN    128

__device__ __forceinline__ bf16_t f2b(float x) { return (bf16_t)x; }

// -------------------- fp32 -> bf16 convert with K padding -------------------
__global__ void cvt_pad_kernel(bf16_t* __restrict__ dst, const float* __restrict__ src,
                               long rows, int K, int KP)
{
    long i = (long)blockIdx.x * blockDim.x + threadIdx.x;
    long n = rows * (long)KP;
    if (i >= n) return;
    long r = i / KP;
    int  k = (int)(i - r * KP);
    dst[i] = (k < K) ? f2b(src[r * (long)K + k]) : (bf16_t)0.0f;
}

// -------------------- combined bias: bih + bhh ------------------------------
__global__ void bias_combine_kernel(float* __restrict__ dst,
                                    const float* __restrict__ a,
                                    const float* __restrict__ b, int n)
{
    int i = blockIdx.x * blockDim.x + threadIdx.x;
    if (i < n) dst[i] = a[i] + b[i];
}

// -------------------- fused dual-GEMM for LSTM gates ------------------------
// G[M=512, N=4096] = A0[M,K0] * W0[N,K0]^T + A1[M,K1] * W1[N,K1]^T + bias[N]
// Wave computes a 32x32 tile (4x v8f accumulators); block = 8 waves (2x4)
// => 64x128 block tile. Per K=32 step: 2 A-frags + 2 B-frags -> 4 WMMAs.
// Fragment layouts used here (contiguous k, same order in A and B):
//   A frag (16x32 bf16): lane = half*16 + m holds A[m][k0+half*16 .. +15]
//   B frag (32x16 bf16): lane = half*16 + n holds W[n][k0+half*16 .. +15]
//   C tile: lane l, vgpr v -> M = v + (l>>4)*8, N = l&15
__global__ __launch_bounds__(256)
void gates_gemm_kernel(float* __restrict__ G,
                       const bf16_t* __restrict__ A0, int lda0,
                       const bf16_t* __restrict__ W0, int K0,
                       const bf16_t* __restrict__ A1, int lda1,
                       const bf16_t* __restrict__ W1, int K1,
                       const float* __restrict__ bias)
{
    const int lane = threadIdx.x & 31;
    const int wave = threadIdx.x >> 5;
    const int wm   = wave >> 2;          // 0..1
    const int wn   = wave & 3;           // 0..3
    const int m0   = blockIdx.y * 64  + wm * 32;
    const int n0   = blockIdx.x * 128 + wn * 32;
    const int r    = lane & 15;          // A-row / B-col within 16-wide subtile
    const int half = lane >> 4;          // which K-half of 32

    v8f acc00 = {}, acc01 = {}, acc10 = {}, acc11 = {};

    {
        const bf16_t* a0 = A0 + (size_t)(m0 + r)      * lda0 + half * 16;
        const bf16_t* a1 = A0 + (size_t)(m0 + 16 + r) * lda0 + half * 16;
        const bf16_t* w0 = W0 + (size_t)(n0 + r)      * K0   + half * 16;
        const bf16_t* w1 = W0 + (size_t)(n0 + 16 + r) * K0   + half * 16;
        for (int k = 0; k < K0; k += 32) {
            v16bf fa0 = *(const v16bf*)(a0 + k);
            v16bf fa1 = *(const v16bf*)(a1 + k);
            v16bf fb0 = *(const v16bf*)(w0 + k);
            v16bf fb1 = *(const v16bf*)(w1 + k);
            // reuse chain: a0 -> a0, b1 -> b1, a1 -> a1
            acc00 = __builtin_amdgcn_wmma_f32_16x16x32_bf16(false, fa0, false, fb0,
                                                            (short)0, acc00, true,  false);
            acc01 = __builtin_amdgcn_wmma_f32_16x16x32_bf16(false, fa0, false, fb1,
                                                            (short)0, acc01, false, true);
            acc11 = __builtin_amdgcn_wmma_f32_16x16x32_bf16(false, fa1, false, fb1,
                                                            (short)0, acc11, true,  false);
            acc10 = __builtin_amdgcn_wmma_f32_16x16x32_bf16(false, fa1, false, fb0,
                                                            (short)0, acc10, false, false);
        }
    }
    {
        const bf16_t* a0 = A1 + (size_t)(m0 + r)      * lda1 + half * 16;
        const bf16_t* a1 = A1 + (size_t)(m0 + 16 + r) * lda1 + half * 16;
        const bf16_t* w0 = W1 + (size_t)(n0 + r)      * K1   + half * 16;
        const bf16_t* w1 = W1 + (size_t)(n0 + 16 + r) * K1   + half * 16;
        for (int k = 0; k < K1; k += 32) {
            v16bf fa0 = *(const v16bf*)(a0 + k);
            v16bf fa1 = *(const v16bf*)(a1 + k);
            v16bf fb0 = *(const v16bf*)(w0 + k);
            v16bf fb1 = *(const v16bf*)(w1 + k);
            acc00 = __builtin_amdgcn_wmma_f32_16x16x32_bf16(false, fa0, false, fb0,
                                                            (short)0, acc00, true,  false);
            acc01 = __builtin_amdgcn_wmma_f32_16x16x32_bf16(false, fa0, false, fb1,
                                                            (short)0, acc01, false, true);
            acc11 = __builtin_amdgcn_wmma_f32_16x16x32_bf16(false, fa1, false, fb1,
                                                            (short)0, acc11, true,  false);
            acc10 = __builtin_amdgcn_wmma_f32_16x16x32_bf16(false, fa1, false, fb0,
                                                            (short)0, acc10, false, false);
        }
    }

    // ---- store 4 sub-tiles with bias ----
    const float bv0 = bias[n0 + r];
    const float bv1 = bias[n0 + 16 + r];
    {
        float* o = G + (size_t)(m0 + half * 8) * NG + n0 + r;
#pragma unroll
        for (int v = 0; v < 8; ++v) {
            o[(size_t)v * NG]      = acc00[v] + bv0;
            o[(size_t)v * NG + 16] = acc01[v] + bv1;
        }
    }
    {
        float* o = G + (size_t)(m0 + 16 + half * 8) * NG + n0 + r;
#pragma unroll
        for (int v = 0; v < 8; ++v) {
            o[(size_t)v * NG]      = acc10[v] + bv0;
            o[(size_t)v * NG + 16] = acc11[v] + bv1;
        }
    }
}

// -------------------- element-wise LSTM cell update -------------------------
// torch gate order i,f,g,o at columns [j, j+H, j+2H, j+3H]
__global__ __launch_bounds__(256)
void cell_update_kernel(const float* __restrict__ G,
                        float* __restrict__ c, float* __restrict__ h,
                        bf16_t* __restrict__ hbf,
                        const int* __restrict__ lengths, int t)
{
    int idx = blockIdx.x * blockDim.x + threadIdx.x;   // b*H + j
    if (idx >= BATCH * HB) return;
    int b = idx >> 10;
    int j = idx & (HB - 1);
    const float* g = G + (size_t)b * NG;
    float gi = g[j], gf = g[j + HB], gg = g[j + 2 * HB], go = g[j + 3 * HB];
    float cp = c[idx], hp = h[idx];

    float si = 1.0f / (1.0f + __expf(-gi));
    float sf = 1.0f / (1.0f + __expf(-gf));
    float so = 1.0f / (1.0f + __expf(-go));
    float cn = sf * cp + si * tanhf(gg);
    float hn = so * tanhf(cn);

    if (lengths != nullptr && t >= lengths[b]) { cn = cp; hn = hp; }

    c[idx]   = cn;
    h[idx]   = hn;
    hbf[idx] = f2b(hn);
}

// -------------------- decoder FC: out = h1 @ fcW^T + fc_b -------------------
// Also writes bf16 feedback (next decoder input). M=512, N=128, K=1024.
__global__ __launch_bounds__(256)
void fc_gemm_kernel(const bf16_t* __restrict__ Hin,   // [512 x 1024]
                    const bf16_t* __restrict__ W,     // [128 x 1024]
                    const float* __restrict__ bias,   // [128]
                    float* __restrict__ out,          // [B, DSTEPS, OUTN]
                    int step,
                    bf16_t* __restrict__ xfeed)       // [B x 128]
{
    const int lane = threadIdx.x & 31;
    const int wave = threadIdx.x >> 5;
    const int wm   = wave >> 2;
    const int wn   = wave & 3;
    const int m0   = blockIdx.y * 32 + wm * 16;
    const int n0   = blockIdx.x * 64 + wn * 16;
    const int r    = lane & 15;
    const int half = lane >> 4;

    v8f acc = {};
    const bf16_t* a = Hin + (size_t)(m0 + r) * HB + half * 16;
    const bf16_t* w = W   + (size_t)(n0 + r) * HB + half * 16;
    for (int k = 0; k < HB; k += 32) {
        v16bf af = *(const v16bf*)(a + k);
        v16bf bf = *(const v16bf*)(w + k);
        acc = __builtin_amdgcn_wmma_f32_16x16x32_bf16(false, af, false, bf,
                                                      (short)0, acc, false, false);
    }

    const float bv = bias[n0 + r];
#pragma unroll
    for (int v = 0; v < 8; ++v) {
        int m = m0 + half * 8 + v;      // batch index
        int n = n0 + r;                 // output channel
        float val = acc[v] + bv;
        out[((size_t)m * DSTEPS + step) * OUTN + n] = val;
        xfeed[(size_t)m * OUTN + n] = f2b(val);
    }
}

// ---------------------------------------------------------------------------
extern "C" void kernel_launch(void* const* d_in, const int* in_sizes, int n_in,
                              void* d_out, int out_size, void* d_ws, size_t ws_size,
                              hipStream_t stream)
{
    (void)in_sizes; (void)n_in; (void)out_size; (void)ws_size;

    const float* x        = (const float*)d_in[0];
    const int*   lengths  = (const int*)  d_in[1];
    const float* eWih0 = (const float*)d_in[2];
    const float* eWhh0 = (const float*)d_in[3];
    const float* ebih0 = (const float*)d_in[4];
    const float* ebhh0 = (const float*)d_in[5];
    const float* eWih1 = (const float*)d_in[6];
    const float* eWhh1 = (const float*)d_in[7];
    const float* ebih1 = (const float*)d_in[8];
    const float* ebhh1 = (const float*)d_in[9];
    const float* dWih0 = (const float*)d_in[10];
    const float* dWhh0 = (const float*)d_in[11];
    const float* dbih0 = (const float*)d_in[12];
    const float* dbhh0 = (const float*)d_in[13];
    const float* dWih1 = (const float*)d_in[14];
    const float* dWhh1 = (const float*)d_in[15];
    const float* dbih1 = (const float*)d_in[16];
    const float* dbhh1 = (const float*)d_in[17];
    const float* fcW   = (const float*)d_in[18];
    const float* fcb   = (const float*)d_in[19];
    float* out = (float*)d_out;

    // ---- workspace bump allocator (256B aligned slices) ----
    char* ws = (char*)d_ws;
    size_t off = 0;
    auto alloc = [&](size_t bytes) -> char* {
        char* p = ws + off;
        off = (off + bytes + 255) & ~(size_t)255;
        return p;
    };

    bf16_t* Xbf  = (bf16_t*)alloc((size_t)BATCH * TSTEPS * KP0 * 2);
    bf16_t* eW0x = (bf16_t*)alloc((size_t)NG * KP0  * 2);
    bf16_t* eW0h = (bf16_t*)alloc((size_t)NG * HB   * 2);
    bf16_t* eW1x = (bf16_t*)alloc((size_t)NG * HB   * 2);
    bf16_t* eW1h = (bf16_t*)alloc((size_t)NG * HB   * 2);
    bf16_t* dW0x = (bf16_t*)alloc((size_t)NG * OUTN * 2);
    bf16_t* dW0h = (bf16_t*)alloc((size_t)NG * HB   * 2);
    bf16_t* dW1x = (bf16_t*)alloc((size_t)NG * HB   * 2);
    bf16_t* dW1h = (bf16_t*)alloc((size_t)NG * HB   * 2);
    bf16_t* fcWb = (bf16_t*)alloc((size_t)OUTN * HB * 2);
    float* bE0   = (float*)alloc((size_t)NG * 4);
    float* bE1   = (float*)alloc((size_t)NG * 4);
    float* bD0   = (float*)alloc((size_t)NG * 4);
    float* bD1   = (float*)alloc((size_t)NG * 4);
    float* G     = (float*)alloc((size_t)BATCH * NG * 4);
    float* h0    = (float*)alloc((size_t)BATCH * HB * 4);
    float* h1    = (float*)alloc((size_t)BATCH * HB * 4);
    float* c0    = (float*)alloc((size_t)BATCH * HB * 4);
    float* c1    = (float*)alloc((size_t)BATCH * HB * 4);
    bf16_t* hb0  = (bf16_t*)alloc((size_t)BATCH * HB * 2);
    bf16_t* hb1  = (bf16_t*)alloc((size_t)BATCH * HB * 2);
    bf16_t* xfeed= (bf16_t*)alloc((size_t)BATCH * OUTN * 2);

    // ---- one-time (per call) weight / input conversion to bf16 ----
    auto cvt = [&](bf16_t* dst, const float* src, long rows, int K, int KP) {
        long n = rows * (long)KP;
        unsigned blocks = (unsigned)((n + 255) / 256);
        cvt_pad_kernel<<<blocks, 256, 0, stream>>>(dst, src, rows, K, KP);
    };
    cvt(Xbf,  x,     (long)BATCH * TSTEPS, DIN,  KP0);   // pad 129 -> 160
    cvt(eW0x, eWih0, NG,   DIN,  KP0);
    cvt(eW0h, eWhh0, NG,   HB,   HB);
    cvt(eW1x, eWih1, NG,   HB,   HB);
    cvt(eW1h, eWhh1, NG,   HB,   HB);
    cvt(dW0x, dWih0, NG,   OUTN, OUTN);
    cvt(dW0h, dWhh0, NG,   HB,   HB);
    cvt(dW1x, dWih1, NG,   HB,   HB);
    cvt(dW1h, dWhh1, NG,   HB,   HB);
    cvt(fcWb, fcW,   OUTN, HB,   HB);

    bias_combine_kernel<<<NG / 256, 256, 0, stream>>>(bE0, ebih0, ebhh0, NG);
    bias_combine_kernel<<<NG / 256, 256, 0, stream>>>(bE1, ebih1, ebhh1, NG);
    bias_combine_kernel<<<NG / 256, 256, 0, stream>>>(bD0, dbih0, dbhh0, NG);
    bias_combine_kernel<<<NG / 256, 256, 0, stream>>>(bD1, dbih1, dbhh1, NG);

    // ---- zero initial state (graph-capture-safe async memsets) ----
    hipMemsetAsync(h0,   0, (size_t)BATCH * HB * 4, stream);
    hipMemsetAsync(h1,   0, (size_t)BATCH * HB * 4, stream);
    hipMemsetAsync(c0,   0, (size_t)BATCH * HB * 4, stream);
    hipMemsetAsync(c1,   0, (size_t)BATCH * HB * 4, stream);
    hipMemsetAsync(hb0,  0, (size_t)BATCH * HB * 2, stream);
    hipMemsetAsync(hb1,  0, (size_t)BATCH * HB * 2, stream);
    hipMemsetAsync(xfeed,0, (size_t)BATCH * OUTN * 2, stream);

    const dim3 ggrid(NG / 128, BATCH / 64);    // 32 x 8 blocks, 256 thr
    const dim3 fgrid(OUTN / 64, BATCH / 32);   // 2 x 16 blocks
    const int  cgrid = (BATCH * HB) / 256;

    // ---- encoder: 82 masked steps ----
    for (int t = 0; t < TSTEPS; ++t) {
        gates_gemm_kernel<<<ggrid, 256, 0, stream>>>(
            G, Xbf + (size_t)t * KP0, TSTEPS * KP0, eW0x, KP0,
            hb0, HB, eW0h, HB, bE0);
        cell_update_kernel<<<cgrid, 256, 0, stream>>>(G, c0, h0, hb0, lengths, t);

        gates_gemm_kernel<<<ggrid, 256, 0, stream>>>(
            G, hb0, HB, eW1x, HB,
            hb1, HB, eW1h, HB, bE1);
        cell_update_kernel<<<cgrid, 256, 0, stream>>>(G, c1, h1, hb1, lengths, t);
    }

    // ---- decoder: 100 autoregressive steps ----
    for (int s = 0; s < DSTEPS; ++s) {
        gates_gemm_kernel<<<ggrid, 256, 0, stream>>>(
            G, xfeed, OUTN, dW0x, OUTN,
            hb0, HB, dW0h, HB, bD0);
        cell_update_kernel<<<cgrid, 256, 0, stream>>>(G, c0, h0, hb0,
                                                      (const int*)nullptr, 0);

        gates_gemm_kernel<<<ggrid, 256, 0, stream>>>(
            G, hb0, HB, dW1x, HB,
            hb1, HB, dW1h, HB, bD1);
        cell_update_kernel<<<cgrid, 256, 0, stream>>>(G, c1, h1, hb1,
                                                      (const int*)nullptr, 0);

        fc_gemm_kernel<<<fgrid, 256, 0, stream>>>(hb1, fcWb, fcb, out, s, xfeed);
    }
}
